// MultiHeadAttention_20023137534765
// MI455X (gfx1250) — compile-verified, matrix operand order checked by reading, not run on hardware
//
#include <hip/hip_runtime.h>
#include <hip/hip_bf16.h>

// ---------------------------------------------------------------------------
// MHA block for MI455X (gfx1250, wave32, WMMA + async LDS copies).
// All GEMMs: bf16 A/B, f32 accumulate via v_wmma_f32_16x16x32_bf16.
// ---------------------------------------------------------------------------

typedef __bf16 bf16x16 __attribute__((ext_vector_type(16)));
typedef float  floatx8 __attribute__((ext_vector_type(8)));

union Frag { bf16x16 v; unsigned short u[16]; };

static constexpr int Bsz = 4, S = 1024, D = 1024, H = 16, DEPTH = 64;

__device__ __forceinline__ unsigned short f32_to_bf16(float x) {
  unsigned int u = __float_as_uint(x);
  unsigned int r = u + 0x7FFFu + ((u >> 16) & 1u);   // round-to-nearest-even
  return (unsigned short)(r >> 16);
}

// Async LDS copy: 16 bytes per lane, tracked on ASYNCcnt (bypasses VGPRs).
// LDS address = low 32 bits of the flat address of a __shared__ element
// (ISA 10.2: flat->LDS truncates addr[31:0] to the LDS offset).
__device__ __forceinline__ void async_copy_b128(void* lds_ptr, const void* gptr) {
  unsigned lds_off = (unsigned)(unsigned long long)lds_ptr;
  asm volatile("global_load_async_to_lds_b128 %0, %1, off"
               :: "v"(lds_off), "v"(gptr)
               : "memory");
}
__device__ __forceinline__ void wait_async0() {
  asm volatile("s_wait_asynccnt 0" ::: "memory");
}

// Load a 16(row) x 32(K) bf16 fragment from an LDS tile stored row-major with
// `stride` elements per row.  Per CDNA5 ISA 16-bit A layout:
//   lane L: row = row0 + (L&15);  VGPR i holds K pair
//   k(i, L>=16) = (i<4?0:16) + 8*(L>>4) + 2*(i&3)
// B fragments use the same pattern with "row" = N (tile stored N-major).
__device__ __forceinline__ bf16x16 load_frag_lds(const unsigned short* lds,
                                                 int stride, int row0, int k0) {
  const int lane = threadIdx.x & 31;
  const int r = row0 + (lane & 15);
  const int hi = lane >> 4;
  Frag f;
#pragma unroll
  for (int i = 0; i < 8; ++i) {
    int k = k0 + ((i & 4) << 2) + (hi << 3) + ((i & 3) << 1);
    f.u[2 * i]     = lds[r * stride + k];
    f.u[2 * i + 1] = lds[r * stride + k + 1];
  }
  return f.v;
}

__device__ __forceinline__ floatx8 wmma_bf16(bf16x16 a, bf16x16 b, floatx8 c) {
  return __builtin_amdgcn_wmma_f32_16x16x32_bf16(
      /*neg_a=*/false, a, /*neg_b=*/false, b,
      /*c_mod=*/(short)0, c, /*reuse_a=*/false, /*reuse_b=*/false);
}

// ---------------------------------------------------------------------------
// Projection GEMM:  Y = X @ W^T + bias   (X fp32 [4096,1024], W fp32 [1024,1024])
// MODE 0: store bf16 head layout [bh][s][d]     (for Q, K)
// MODE 1: store bf16 transposed  [bh][d][s]     (for V -> contiguous B pairs)
// Block = 256 thr (8 waves); tile 128(M) x 64(N); BK = 32.
// Both tiles need fp32->bf16 conversion, so staging stays in VGPRs here.
// ---------------------------------------------------------------------------
template <int MODE>
__global__ __launch_bounds__(256) void proj_kernel(
    const float* __restrict__ X, const float* __restrict__ W,
    const float* __restrict__ bias, unsigned short* __restrict__ out) {
  __shared__ unsigned short As[128 * 32];
  __shared__ unsigned short Bs[64 * 32];
  const int m0 = blockIdx.y * 128;
  const int n0 = blockIdx.x * 64;
  const int tid = threadIdx.x, wave = tid >> 5, lane = tid & 31;

  floatx8 zf = {};
  floatx8 acc[4] = {zf, zf, zf, zf};

  for (int k0 = 0; k0 < D; k0 += 32) {
    if (k0 + 32 < D) {  // prefetch next A tile slab -> global_prefetch_b8
      __builtin_prefetch(&X[(m0 + (tid >> 1)) * D + k0 + 32], 0, 1);
    }
#pragma unroll 4
    for (int i = tid; i < 128 * 32; i += 256) {
      int r = i >> 5, c = i & 31;
      As[i] = f32_to_bf16(X[(m0 + r) * D + k0 + c]);
    }
#pragma unroll 2
    for (int i = tid; i < 64 * 32; i += 256) {
      int r = i >> 5, c = i & 31;
      Bs[i] = f32_to_bf16(W[(n0 + r) * D + k0 + c]);
    }
    __syncthreads();
    bf16x16 a = load_frag_lds(As, 32, wave * 16, 0);
#pragma unroll
    for (int nt = 0; nt < 4; ++nt) {
      bf16x16 b = load_frag_lds(Bs, 32, nt * 16, 0);
      acc[nt] = wmma_bf16(a, b, acc[nt]);
    }
    __syncthreads();
  }

  const int hi = lane >> 4;
#pragma unroll
  for (int nt = 0; nt < 4; ++nt) {
#pragma unroll
    for (int r = 0; r < 8; ++r) {
      int row = m0 + wave * 16 + r + 8 * hi;   // 0..4095  (b*1024+s)
      int col = n0 + nt * 16 + (lane & 15);    // 0..1023  (h*64+d)
      float v = acc[nt][r] + bias[col];
      int bb = row >> 10, s = row & 1023;
      int hd = col >> 6,  d = col & 63;
      int bh = bb * H + hd;
      long idx = (MODE == 0) ? ((long)(bh * S + s) * DEPTH + d)
                             : ((long)(bh * DEPTH + d) * S + s);
      out[idx] = f32_to_bf16(v);
    }
  }
}

// ---------------------------------------------------------------------------
// Logits:  attn_raw[bh,q,k] = (Qh @ Kh^T)/8 - 1e9*mask[b,k] + adjoin[b,q,k]
// Block tile: 128(q) x 64(key), full depth (two K=32 WMMA steps).
// Q/K tiles are bf16 already -> async LDS copy (ASYNCcnt path).
// ---------------------------------------------------------------------------
__global__ __launch_bounds__(256) void logits_kernel(
    const unsigned short* __restrict__ qh, const unsigned short* __restrict__ kh,
    const float* __restrict__ mask, const float* __restrict__ adjoin,
    float* __restrict__ attn) {
  __shared__ unsigned short Qs[128 * 64];   // 16 KB
  __shared__ unsigned short Ks[64 * 64];    //  8 KB
  const int bh = blockIdx.z;
  const int q0 = blockIdx.y * 128;
  const int n0 = blockIdx.x * 64;
  const int bb = bh >> 4;
  const int tid = threadIdx.x, wave = tid >> 5, lane = tid & 31;

  const unsigned short* qb = qh + (long)bh * S * DEPTH;
  const unsigned short* kb = kh + (long)bh * S * DEPTH;
  // 16-byte chunks: 8 per 64-element row.
#pragma unroll
  for (int i = tid; i < 128 * 8; i += 256) {
    int r = i >> 3, c = (i & 7) * 8;
    async_copy_b128(&Qs[r * 64 + c], qb + (q0 + r) * 64 + c);
  }
#pragma unroll
  for (int i = tid; i < 64 * 8; i += 256) {
    int r = i >> 3, c = (i & 7) * 8;
    async_copy_b128(&Ks[r * 64 + c], kb + (n0 + r) * 64 + c);
  }
  wait_async0();
  __syncthreads();

  floatx8 zf = {};
  floatx8 acc[4] = {zf, zf, zf, zf};
#pragma unroll
  for (int ks = 0; ks < DEPTH; ks += 32) {
    bf16x16 a = load_frag_lds(Qs, DEPTH, wave * 16, ks);
#pragma unroll
    for (int nt = 0; nt < 4; ++nt) {
      bf16x16 b = load_frag_lds(Ks, DEPTH, nt * 16, ks);
      acc[nt] = wmma_bf16(a, b, acc[nt]);
    }
  }

  const int hi = lane >> 4;
#pragma unroll
  for (int nt = 0; nt < 4; ++nt) {
#pragma unroll
    for (int r = 0; r < 8; ++r) {
      int qg = q0 + wave * 16 + r + 8 * hi;
      int kg = n0 + nt * 16 + (lane & 15);
      float v = acc[nt][r] * 0.125f
              + mask[bb * S + kg] * -1e9f
              + adjoin[((long)bb * S + qg) * S + kg];
      attn[((long)bh * S + qg) * S + kg] = v;
    }
  }
}

// ---------------------------------------------------------------------------
// Row softmax in place: 65536 rows x 1024, one 256-thread block per row.
// ---------------------------------------------------------------------------
__global__ __launch_bounds__(256) void softmax_kernel(float* __restrict__ attn) {
  __shared__ float red[256];
  float* p = attn + (long)blockIdx.x * S;
  const int tid = threadIdx.x;
  float x[4];
  float m = -3.402823466e38f;
#pragma unroll
  for (int j = 0; j < 4; ++j) { x[j] = p[tid + 256 * j]; m = fmaxf(m, x[j]); }
  red[tid] = m;
  __syncthreads();
  for (int s = 128; s > 0; s >>= 1) {
    if (tid < s) red[tid] = fmaxf(red[tid], red[tid + s]);
    __syncthreads();
  }
  m = red[0];
  __syncthreads();
  float sum = 0.f;
#pragma unroll
  for (int j = 0; j < 4; ++j) { x[j] = __expf(x[j] - m); sum += x[j]; }
  red[tid] = sum;
  __syncthreads();
  for (int s = 128; s > 0; s >>= 1) {
    if (tid < s) red[tid] += red[tid + s];
    __syncthreads();
  }
  float inv = 1.0f / red[0];
#pragma unroll
  for (int j = 0; j < 4; ++j) p[tid + 256 * j] = x[j] * inv;
}

// ---------------------------------------------------------------------------
// ctx = attn @ Vh  per (b,h); writes ctx bf16 in [b, s, h*64+d] flat layout.
// Block tile: 128(q) x 64(depth); K loop over 1024 keys in steps of 32.
// V tile is bf16 already -> async LDS copy overlapping the fp32 attn staging.
// ---------------------------------------------------------------------------
__global__ __launch_bounds__(256) void ctx_kernel(
    const float* __restrict__ attn, const unsigned short* __restrict__ vt,
    unsigned short* __restrict__ ctx) {
  __shared__ unsigned short As[128 * 32];
  __shared__ unsigned short Bs[64 * 32];
  const int bh = blockIdx.y;
  const int q0 = blockIdx.x * 128;
  const int bb = bh >> 4, hd = bh & 15;
  const int tid = threadIdx.x, wave = tid >> 5, lane = tid & 31;
  const float* ab = attn + (long)bh * S * S;
  const unsigned short* vb = vt + (long)bh * DEPTH * S;

  floatx8 zf = {};
  floatx8 acc[4] = {zf, zf, zf, zf};

  for (int k0 = 0; k0 < S; k0 += 32) {
    {  // async copy of the V tile: 64 rows x 32 cols = 256 x 16B chunks
      int r = tid >> 2, c = (tid & 3) * 8;
      async_copy_b128(&Bs[r * 32 + c], vb + r * S + k0 + c);
    }
#pragma unroll 4
    for (int i = tid; i < 128 * 32; i += 256) {
      int r = i >> 5, c = i & 31;
      As[i] = f32_to_bf16(ab[(long)(q0 + r) * S + k0 + c]);
    }
    wait_async0();
    __syncthreads();
    bf16x16 a = load_frag_lds(As, 32, wave * 16, 0);
#pragma unroll
    for (int nt = 0; nt < 4; ++nt) {
      bf16x16 b = load_frag_lds(Bs, 32, nt * 16, 0);
      acc[nt] = wmma_bf16(a, b, acc[nt]);
    }
    __syncthreads();
  }

  const int hi = lane >> 4;
#pragma unroll
  for (int nt = 0; nt < 4; ++nt) {
#pragma unroll
    for (int r = 0; r < 8; ++r) {
      int qg = q0 + wave * 16 + r + 8 * hi;
      int d  = nt * 16 + (lane & 15);
      ctx[((long)(bb * S + qg)) * D + hd * DEPTH + d] = f32_to_bf16(acc[nt][r]);
    }
  }
}

// ---------------------------------------------------------------------------
// Output dense: out = ctx(bf16) @ dense_w^T + dense_b  -> fp32 d_out[0:4M)
// ctx tile is bf16 already -> async LDS copy; W tile converts via VGPRs.
// ---------------------------------------------------------------------------
__global__ __launch_bounds__(256) void dense_kernel(
    const unsigned short* __restrict__ Xbf, const float* __restrict__ W,
    const float* __restrict__ bias, float* __restrict__ out) {
  __shared__ unsigned short As[128 * 32];
  __shared__ unsigned short Bs[64 * 32];
  const int m0 = blockIdx.y * 128;
  const int n0 = blockIdx.x * 64;
  const int tid = threadIdx.x, wave = tid >> 5, lane = tid & 31;

  floatx8 zf = {};
  floatx8 acc[4] = {zf, zf, zf, zf};

  for (int k0 = 0; k0 < D; k0 += 32) {
#pragma unroll
    for (int i = tid; i < 128 * 4; i += 256) {   // 512 x 16B chunks
      int r = i >> 2, c = (i & 3) * 8;
      async_copy_b128(&As[r * 32 + c], Xbf + (m0 + r) * D + k0 + c);
    }
#pragma unroll 2
    for (int i = tid; i < 64 * 32; i += 256) {
      int r = i >> 5, c = i & 31;
      Bs[i] = f32_to_bf16(W[(n0 + r) * D + k0 + c]);
    }
    wait_async0();
    __syncthreads();
    bf16x16 a = load_frag_lds(As, 32, wave * 16, 0);
#pragma unroll
    for (int nt = 0; nt < 4; ++nt) {
      bf16x16 b = load_frag_lds(Bs, 32, nt * 16, 0);
      acc[nt] = wmma_bf16(a, b, acc[nt]);
    }
    __syncthreads();
  }

  const int hi = lane >> 4;
#pragma unroll
  for (int nt = 0; nt < 4; ++nt) {
#pragma unroll
    for (int r = 0; r < 8; ++r) {
      int row = m0 + wave * 16 + r + 8 * hi;
      int col = n0 + nt * 16 + (lane & 15);
      out[(long)row * D + col] = acc[nt][r] + bias[col];
    }
  }
}

// ---------------------------------------------------------------------------
extern "C" void kernel_launch(void* const* d_in, const int* in_sizes, int n_in,
                              void* d_out, int out_size, void* d_ws, size_t ws_size,
                              hipStream_t stream) {
  const float* q       = (const float*)d_in[0];
  const float* k       = (const float*)d_in[1];
  const float* v       = (const float*)d_in[2];
  const float* mask    = (const float*)d_in[3];
  const float* adjoin  = (const float*)d_in[4];
  const float* wq_w    = (const float*)d_in[5];
  const float* wq_b    = (const float*)d_in[6];
  const float* wk_w    = (const float*)d_in[7];
  const float* wk_b    = (const float*)d_in[8];
  const float* wv_w    = (const float*)d_in[9];
  const float* wv_b    = (const float*)d_in[10];
  const float* dense_w = (const float*)d_in[11];
  const float* dense_b = (const float*)d_in[12];

  float* out  = (float*)d_out;                       // [B,S,D] = 4,194,304 f32
  float* attn = out + (long)Bsz * S * D;             // [B,H,S,S] fp32

  // Workspace: 4 x 8 MB bf16 buffers = 32 MB
  unsigned short* qh  = (unsigned short*)d_ws;       // [bh][s][d]
  unsigned short* kh  = qh + (long)Bsz * S * D;      // [bh][s][d]
  unsigned short* vt  = kh + (long)Bsz * S * D;      // [bh][d][s]
  unsigned short* ctx = vt + (long)Bsz * S * D;      // [b*s][h*64+d]

  dim3 blk(256);
  proj_kernel<0><<<dim3(D / 64, (Bsz * S) / 128), blk, 0, stream>>>(q, wq_w, wq_b, qh);
  proj_kernel<0><<<dim3(D / 64, (Bsz * S) / 128), blk, 0, stream>>>(k, wk_w, wk_b, kh);
  proj_kernel<1><<<dim3(D / 64, (Bsz * S) / 128), blk, 0, stream>>>(v, wv_w, wv_b, vt);

  logits_kernel<<<dim3(S / 64, S / 128, Bsz * H), blk, 0, stream>>>(qh, kh, mask, adjoin, attn);
  softmax_kernel<<<dim3(Bsz * H * S), blk, 0, stream>>>(attn);
  ctx_kernel<<<dim3(S / 128, Bsz * H), blk, 0, stream>>>(attn, vt, ctx);
  dense_kernel<<<dim3(D / 64, (Bsz * S) / 128), blk, 0, stream>>>(ctx, dense_w, dense_b, out);
}
